// JBULearnedRange_26989574488719
// MI455X (gfx1250) — matrix operand order repaired and checked by reading, not compiled, not run on hardware
//
#include <hip/hip_runtime.h>
#include <hip/hip_bf16.h>

typedef __attribute__((ext_vector_type(16))) __bf16 v16bf;
typedef __attribute__((ext_vector_type(8)))  __bf16 bf16x8;
typedef __attribute__((ext_vector_type(8)))  float  v8f;

#define GHh 224
#define GWw 224
#define SHh 112
#define SWw 112
#define GCc 384
#define KD  32
#define D2  49
#define NPIX (2 * GHh * GWw)

__device__ __forceinline__ float geluf(float x) {
    return 0.5f * x * (1.0f + erff(x * 0.70710678118654752f));
}
__device__ __forceinline__ int reflect224(int i) {
    i = (i < 0) ? -i : i;
    return (i > 223) ? (446 - i) : i;
}
__device__ __forceinline__ float cubw(float s) {
    const float a = -0.75f;
    s = fabsf(s);
    if (s <= 1.0f) return ((a + 2.0f) * s - (a + 3.0f)) * s * s + 1.0f;
    if (s <  2.0f) return ((a * s - 5.0f * a) * s + 8.0f * a) * s - 4.0f * a;
    return 0.0f;
}
__device__ __forceinline__ v8f wmma_bf16(v16bf a, v16bf b, v8f c) {
    return __builtin_amdgcn_wmma_f32_16x16x32_bf16(false, a, false, b, (short)0, c, false, false);
}
// A-fragment K index for 16-bit 16x32 A (contiguous): element e, lane half h -> K = h*16 + e
__device__ __forceinline__ int a_kidx(int e, int half) {
    return e + (half ? 16 : 0);
}
// load a 16-element bf16 fragment stored contiguously (pre-packed layout)
__device__ __forceinline__ v16bf load_frag16(const __bf16* p) {
    bf16x8 lo = *(const bf16x8*)p;
    bf16x8 hi = *(const bf16x8*)(p + 8);
    v16bf f;
    #pragma unroll
    for (int e = 0; e < 8; ++e) { f[e] = lo[e]; f[e + 8] = hi[e]; }
    return f;
}

// fragment-packed weight buffer sizes (elements)
#define W1F_N (2 * 12 * 32 * 16)   // 12288
#define W2F_N (2 * 32 * 16)        // 1024
#define FW1F_N (4 * 14 * 32 * 16)  // 28672
#define FW2F_N (4 * 2 * 32 * 16)   // 4096

// ---- kw: pre-pack all weight matrices into WMMA B-fragment order, bf16 ----
__global__ __launch_bounds__(256) void kw_prep(const float* __restrict__ w1,
                                               const float* __restrict__ w2,
                                               const float* __restrict__ fw1,
                                               const float* __restrict__ fw2,
                                               __bf16* __restrict__ w1f,
                                               __bf16* __restrict__ w2f,
                                               __bf16* __restrict__ fw1f,
                                               __bf16* __restrict__ fw2f) {
    // B-fragment: lane half h, lane index l (=N), element e -> K = h*16 + e (within chunk)
    for (int idx = threadIdx.x; idx < W1F_N; idx += 256) {
        const int e = idx & 15; int t = idx >> 4;
        const int lane = t & 31; t >>= 5;
        const int kc = t % 12, nt = t / 12;
        const int half = lane >> 4, lidx = lane & 15;
        w1f[idx] = (__bf16)w1[(nt * 16 + lidx) * GCc + kc * 32 + half * 16 + e];
    }
    for (int idx = threadIdx.x; idx < W2F_N; idx += 256) {
        const int e = idx & 15; int t = idx >> 4;
        const int lane = t & 31;
        const int nt = t >> 5;
        const int half = lane >> 4, lidx = lane & 15;
        w2f[idx] = (__bf16)w2[(nt * 16 + lidx) * KD + half * 16 + e];
    }
    for (int idx = threadIdx.x; idx < FW1F_N; idx += 256) {
        const int e = idx & 15; int t = idx >> 4;
        const int lane = t & 31; t >>= 5;
        const int kc = t % 14, nt = t / 14;
        const int half = lane >> 4, lidx = lane & 15;
        const int n = nt * 16 + lidx;
        const int gk = kc * 32 + half * 16 + e;
        fw1f[idx] = (__bf16)((n < D2 && gk < D2 + GCc) ? fw1[n * (D2 + GCc) + gk] : 0.f);
    }
    for (int idx = threadIdx.x; idx < FW2F_N; idx += 256) {
        const int e = idx & 15; int t = idx >> 4;
        const int lane = t & 31; t >>= 5;
        const int c2 = t & 1, nt = t >> 1;
        const int half = lane >> 4, lidx = lane & 15;
        const int n = nt * 16 + lidx;
        const int gk = c2 * 32 + half * 16 + e;
        fw2f[idx] = (__bf16)((n < D2 && gk < D2) ? fw2[n * D2 + gk] : 0.f);
    }
}

// ---------------- k0: guidance [B,C,H,W] f32 -> g_t [B,H,W,C] bf16 ----------------
__global__ __launch_bounds__(256) void k0_transpose(const float* __restrict__ g,
                                                    __bf16* __restrict__ gt) {
    __shared__ float tile[32][33];
    const int wt = blockIdx.x * 32;
    const int h  = blockIdx.y;
    const int bz = blockIdx.z;
    const int b  = bz / 12;
    const int ct = (bz % 12) * 32;
    const int tx = threadIdx.x & 31;
    const int ty = threadIdx.x >> 5;
    #pragma unroll
    for (int i = ty; i < 32; i += 8)
        tile[i][tx] = g[(((size_t)b * GCc + ct + i) * GHh + h) * GWw + wt + tx];
    __syncthreads();
    #pragma unroll
    for (int i = ty; i < 32; i += 8)
        gt[(((size_t)b * GHh + h) * GWw + wt + i) * GCc + ct + tx] = (__bf16)tile[tx][i];
}

// ---------------- k1: proj = conv1x1(gelu(conv1x1(g))) via WMMA ----------------
__global__ __launch_bounds__(256) void k1_proj(const __bf16* __restrict__ gt,
                                               const __bf16* __restrict__ w1f,
                                               const float* __restrict__ b1,
                                               const __bf16* __restrict__ w2f,
                                               const float* __restrict__ b2,
                                               float* __restrict__ proj) {
    __shared__ float hid[8][16][KD];
    const int wave = threadIdx.x >> 5;
    const int lane = threadIdx.x & 31;
    const int half = lane >> 4;
    const int lidx = lane & 15;
    const int pix0 = (blockIdx.x * 8 + wave) * 16;

    v8f acc[2];
    #pragma unroll
    for (int nt = 0; nt < 2; ++nt) {
        const float bias = b1[nt * 16 + lidx];
        #pragma unroll
        for (int r = 0; r < 8; ++r) acc[nt][r] = bias;
    }

    const __bf16* arow = gt + (size_t)(pix0 + lidx) * GCc + (half ? 16 : 0);

    #pragma unroll
    for (int ci = 0; ci < 12; ++ci) {
        const int kc = ci * 32;
        v16bf afrag;
        bf16x8 lo = *(const bf16x8*)(arow + kc);
        bf16x8 hi = *(const bf16x8*)(arow + kc + 8);
        #pragma unroll
        for (int e = 0; e < 8; ++e) { afrag[e] = lo[e]; afrag[e + 8] = hi[e]; }
        #pragma unroll
        for (int nt = 0; nt < 2; ++nt) {
            v16bf bfrag = load_frag16(w1f + ((nt * 12 + ci) * 32 + lane) * 16);
            acc[nt] = wmma_bf16(afrag, bfrag, acc[nt]);
        }
    }
    #pragma unroll
    for (int nt = 0; nt < 2; ++nt)
        #pragma unroll
        for (int r = 0; r < 8; ++r)
            hid[wave][r + half * 8][nt * 16 + lidx] = geluf(acc[nt][r]);
    __syncthreads();

    v16bf a2;
    #pragma unroll
    for (int e = 0; e < 16; ++e) a2[e] = (__bf16)hid[wave][lidx][a_kidx(e, half)];

    #pragma unroll
    for (int nt = 0; nt < 2; ++nt) {
        v8f acc2;
        const float bias = b2[nt * 16 + lidx];
        #pragma unroll
        for (int r = 0; r < 8; ++r) acc2[r] = bias;
        v16bf bfrag = load_frag16(w2f + (nt * 32 + lane) * 16);
        acc2 = wmma_bf16(a2, bfrag, acc2);
        #pragma unroll
        for (int r = 0; r < 8; ++r)
            proj[(size_t)(pix0 + r + half * 8) * KD + nt * 16 + lidx] = acc2[r];
    }
}

// ------- k2: scores (49 neighbor dots) + softmax + spatial + normalize -------
__global__ __launch_bounds__(256) void k2_combined(const float* __restrict__ proj,
                                                   const float* __restrict__ rt_p,
                                                   const float* __restrict__ ss_p,
                                                   float* __restrict__ comb) {
    __shared__ float pt[22 * 22 * KD];   // 61952 B: proj halo tile
    const int b  = blockIdx.z;
    const int H0 = blockIdx.y * 16, W0 = blockIdx.x * 16;
    for (int idx = threadIdx.x; idx < 22 * 22 * KD; idx += 256) {
        const int k  = idx & 31;
        const int pp = idx >> 5;
        const int lh = pp / 22, lw = pp % 22;
        const int gh = reflect224(H0 - 3 + lh);
        const int gw = reflect224(W0 - 3 + lw);
        pt[idx] = proj[(((size_t)b * GHh + gh) * GWw + gw) * KD + k];
    }
    __syncthreads();
    const int ty = threadIdx.x >> 4, tx = threadIdx.x & 15;
    float ctr[KD];
    const float* cp = &pt[((ty + 3) * 22 + (tx + 3)) * KD];
    #pragma unroll
    for (int k = 0; k < KD; ++k) ctr[k] = cp[k];

    float sc[D2];
    float mx = -3.4e38f;
    #pragma unroll
    for (int di = 0; di < 7; ++di)
        #pragma unroll
        for (int dj = 0; dj < 7; ++dj) {
            const float* nb = &pt[((ty + di) * 22 + (tx + dj)) * KD];
            float s = 0.f;
            #pragma unroll
            for (int k = 0; k < KD; ++k) s += nb[k] * ctr[k];
            sc[di * 7 + dj] = s;
            mx = fmaxf(mx, s);
        }
    const float temp = fminf(fmaxf(expf(rt_p[0]), 1e-4f), 1e4f);
    float ssum = 0.f;
    #pragma unroll
    for (int p = 0; p < D2; ++p) { sc[p] = expf(temp * (sc[p] - mx)); ssum += sc[p]; }
    const float sig  = ss_p[0];
    const float inv2 = 1.0f / (2.0f * sig * sig);
    const float isum = 1.0f / ssum;
    float csum = 0.f;
    #pragma unroll
    for (int di = 0; di < 7; ++di) {
        const float xx = (float)(di - 3) * (1.0f / 3.0f);
        #pragma unroll
        for (int dj = 0; dj < 7; ++dj) {
            const float yy = (float)(dj - 3) * (1.0f / 3.0f);
            const float sp = expf(-(xx * xx + yy * yy) * inv2);
            const int p = di * 7 + dj;
            sc[p] = sc[p] * isum * sp;
            csum += sc[p];
        }
    }
    const float inv = 1.0f / fmaxf(csum, 1e-7f);
    const size_t pix = ((size_t)b * GHh + H0 + ty) * GWw + W0 + tx;
    #pragma unroll
    for (int p = 0; p < D2; ++p) comb[pix * D2 + p] = sc[p] * inv;
}

// ------- k3: fixup MLP (433->49 gelu 49->49) via WMMA, kern = comb + 0.1*fixup -------
__global__ __launch_bounds__(256) void k3_fixup(const __bf16* __restrict__ gt,
                                                const float* __restrict__ comb,
                                                const __bf16* __restrict__ fw1f,
                                                const float* __restrict__ fb1,
                                                const __bf16* __restrict__ fw2f,
                                                const float* __restrict__ fb2,
                                                float* __restrict__ kern) {
    __shared__ float hid[8][16][64];
    const int wave = threadIdx.x >> 5;
    const int lane = threadIdx.x & 31;
    const int half = lane >> 4;
    const int lidx = lane & 15;
    const int pix0 = (blockIdx.x * 8 + wave) * 16;

    v8f acc[4];
    #pragma unroll
    for (int nt = 0; nt < 4; ++nt) {
        const int n = nt * 16 + lidx;
        const float bias = (n < D2) ? fb1[n] : 0.f;
        #pragma unroll
        for (int r = 0; r < 8; ++r) acc[nt][r] = bias;
    }

    const float*  crow = comb + (size_t)(pix0 + lidx) * D2;
    const __bf16* grow = gt   + (size_t)(pix0 + lidx) * GCc;

    // K = 49 (combined) + 384 (guidance) = 433, padded to 448 = 14 chunks of 32.
    // Fully unrolled so region tests constant-fold: chunks >= 2 are pure bf16 loads.
    #pragma unroll
    for (int ci = 0; ci < 14; ++ci) {
        const int kc = ci * 32;
        v16bf afrag;
        #pragma unroll
        for (int e = 0; e < 16; ++e) {
            const int gk = kc + a_kidx(e, half);
            __bf16 v;
            if (gk < D2)            v = (__bf16)crow[gk];
            else if (gk < D2 + GCc) v = grow[gk - D2];
            else                    v = (__bf16)0.f;
            afrag[e] = v;
        }
        #pragma unroll
        for (int nt = 0; nt < 4; ++nt) {
            v16bf bfrag = load_frag16(fw1f + ((nt * 14 + ci) * 32 + lane) * 16);
            acc[nt] = wmma_bf16(afrag, bfrag, acc[nt]);
        }
    }
    #pragma unroll
    for (int nt = 0; nt < 4; ++nt)
        #pragma unroll
        for (int r = 0; r < 8; ++r)
            hid[wave][r + half * 8][nt * 16 + lidx] = geluf(acc[nt][r]);
    __syncthreads();

    v16bf a2[2];
    #pragma unroll
    for (int c2 = 0; c2 < 2; ++c2)
        #pragma unroll
        for (int e = 0; e < 16; ++e)
            a2[c2][e] = (__bf16)hid[wave][lidx][c2 * 32 + a_kidx(e, half)];

    #pragma unroll
    for (int nt = 0; nt < 4; ++nt) {
        const int n = nt * 16 + lidx;
        v8f acc2;
        const float bias = (n < D2) ? fb2[n] : 0.f;
        #pragma unroll
        for (int r = 0; r < 8; ++r) acc2[r] = bias;
        #pragma unroll
        for (int c2 = 0; c2 < 2; ++c2) {
            v16bf bfrag = load_frag16(fw2f + ((nt * 2 + c2) * 32 + lane) * 16);
            acc2 = wmma_bf16(a2[c2], bfrag, acc2);
        }
        if (n < D2) {
            #pragma unroll
            for (int r = 0; r < 8; ++r) {
                const size_t o = (size_t)(pix0 + r + half * 8) * D2 + n;
                kern[o] = comb[o] + 0.1f * acc2[r];
            }
        }
    }
}

// ------- k5: fused bicubic x2 upsample + 49-tap adaptive conv (384 channels) -------
__global__ __launch_bounds__(256) void k5_upconv(const float* __restrict__ src,
                                                 const float* __restrict__ kern,
                                                 float* __restrict__ out) {
    __shared__ float kls[16 * 16 * D2];   // 50176 B
    __shared__ float sls[14 * 14];
    __shared__ float tmp[14][22];
    __shared__ float hr[22][22];
    __shared__ float wrow[22][4], wcol[22][4];
    __shared__ int   irow[22][4], icol[22][4];

    const int bz = blockIdx.z;
    const int b = bz >> 2, cg = bz & 3;
    const int H0 = blockIdx.y * 16, W0 = blockIdx.x * 16;

    for (int idx = threadIdx.x; idx < 16 * 16 * D2; idx += 256) {
        const int q = idx % D2, p = idx / D2;
        const int py = p >> 4, px = p & 15;
        kls[idx] = kern[(((size_t)b * GHh + H0 + py) * GWw + W0 + px) * D2 + q];
    }
    if (threadIdx.x < 44) {
        const int r = (int)threadIdx.x % 22;
        const bool isRow = threadIdx.x < 22;
        const int G0 = isRow ? H0 : W0;
        const int g = reflect224(G0 - 3 + r);   // reflect-pad of hr == evaluate at reflected coord
        const float x  = (float)g * 0.5f - 0.25f;
        const float xf = floorf(x);
        const int   x0 = (int)xf;
        const float t  = x - xf;
        const int base = (G0 >> 1) - 3;
        #pragma unroll
        for (int k = 0; k < 4; ++k) {
            const float w = cubw(t - (float)(k - 1));
            int ii = x0 + k - 1;
            ii = (ii < 0) ? 0 : ((ii > SHh - 1) ? SHh - 1 : ii);
            ii -= base;
            if (isRow) { wrow[r][k] = w; irow[r][k] = ii; }
            else       { wcol[r][k] = w; icol[r][k] = ii; }
        }
    }
    __syncthreads();
    const int ty = threadIdx.x >> 4, tx = threadIdx.x & 15;
    const int srb = (H0 >> 1) - 3, scb = (W0 >> 1) - 3;
    const int c0 = cg * 96;

    for (int c = c0; c < c0 + 96; ++c) {
        const float* sp = src + ((size_t)b * GCc + c) * SHh * SWw;
        // prefetch stays inside the allocation: last channel prefetches itself
        const float* spn = (c + 1 < c0 + 96) ? (sp + SHh * SWw) : sp;
        for (int idx = threadIdx.x; idx < 196; idx += 256) {
            const int r = idx / 14, cc = idx % 14;
            int gr = srb + r;  gr = (gr < 0) ? 0 : ((gr > SHh - 1) ? SHh - 1 : gr);
            int gc = scb + cc; gc = (gc < 0) ? 0 : ((gc > SWw - 1) ? SWw - 1 : gc);
            sls[idx] = sp[gr * SWw + gc];
            __builtin_prefetch(spn + gr * SWw + gc, 0, 1);   // global_prefetch_b8
        }
        __syncthreads();
        for (int idx = threadIdx.x; idx < 14 * 22; idx += 256) {
            const int r = idx / 22, cc = idx % 22;
            float s = 0.f;
            #pragma unroll
            for (int k = 0; k < 4; ++k) s += wcol[cc][k] * sls[r * 14 + icol[cc][k]];
            tmp[r][cc] = s;
        }
        __syncthreads();
        for (int idx = threadIdx.x; idx < 22 * 22; idx += 256) {
            const int rr = idx / 22, cc = idx % 22;
            float s = 0.f;
            #pragma unroll
            for (int k = 0; k < 4; ++k) s += wrow[rr][k] * tmp[irow[rr][k]][cc];
            hr[rr][cc] = s;
        }
        __syncthreads();
        const float* kp = &kls[(ty * 16 + tx) * D2];
        float o = 0.f;
        #pragma unroll
        for (int di = 0; di < 7; ++di)
            #pragma unroll
            for (int dj = 0; dj < 7; ++dj)
                o += hr[ty + di][tx + dj] * kp[di * 7 + dj];
        out[(((size_t)b * GCc + c) * GHh + H0 + ty) * GWw + W0 + tx] = o;
        __syncthreads();
    }
}

extern "C" void kernel_launch(void* const* d_in, const int* in_sizes, int n_in,
                              void* d_out, int out_size, void* d_ws, size_t ws_size,
                              hipStream_t stream) {
    (void)in_sizes; (void)n_in; (void)out_size; (void)ws_size;
    const float* src   = (const float*)d_in[0];
    const float* gdn   = (const float*)d_in[1];
    const float* rt    = (const float*)d_in[2];
    const float* ss    = (const float*)d_in[3];
    const float* rp_w1 = (const float*)d_in[4];
    const float* rp_b1 = (const float*)d_in[5];
    const float* rp_w2 = (const float*)d_in[6];
    const float* rp_b2 = (const float*)d_in[7];
    const float* fx_w1 = (const float*)d_in[8];
    const float* fx_b1 = (const float*)d_in[9];
    const float* fx_w2 = (const float*)d_in[10];
    const float* fx_b2 = (const float*)d_in[11];

    char* ws = (char*)d_ws;
    size_t off = 0;
    __bf16* gt   = (__bf16*)(ws + off); off += (size_t)NPIX * GCc * 2; off = (off + 255) & ~(size_t)255;
    float*  proj = (float*)(ws + off);  off += (size_t)NPIX * KD * 4;  off = (off + 255) & ~(size_t)255;
    float*  comb = (float*)(ws + off);  off += (size_t)NPIX * D2 * 4;  off = (off + 255) & ~(size_t)255;
    float*  kern = (float*)(ws + off);  off += (size_t)NPIX * D2 * 4;  off = (off + 255) & ~(size_t)255;
    __bf16* w1f  = (__bf16*)(ws + off); off += (size_t)W1F_N * 2;      off = (off + 255) & ~(size_t)255;
    __bf16* w2f  = (__bf16*)(ws + off); off += (size_t)W2F_N * 2;      off = (off + 255) & ~(size_t)255;
    __bf16* fw1f = (__bf16*)(ws + off); off += (size_t)FW1F_N * 2;     off = (off + 255) & ~(size_t)255;
    __bf16* fw2f = (__bf16*)(ws + off);

    kw_prep<<<dim3(1), 256, 0, stream>>>(rp_w1, rp_w2, fx_w1, fx_w2, w1f, w2f, fw1f, fw2f);
    k0_transpose<<<dim3(7, 224, 24), 256, 0, stream>>>(gdn, gt);
    k1_proj<<<dim3(NPIX / 16 / 8), 256, 0, stream>>>(gt, w1f, rp_b1, w2f, rp_b2, proj);
    k2_combined<<<dim3(14, 14, 2), 256, 0, stream>>>(proj, rt, ss, comb);
    k3_fixup<<<dim3(NPIX / 16 / 8), 256, 0, stream>>>(gt, comb, fw1f, fx_b1, fw2f, fx_b2, kern);
    k5_upconv<<<dim3(14, 14, 8), 256, 0, stream>>>(src, kern, (float*)d_out);
}
